// CrossModalMultiHeadAttentionK_70987219469039
// MI455X (gfx1250) — matrix operand order repaired and checked by reading, not compiled, not run on hardware
//
#include <hip/hip_runtime.h>
#include <hip/hip_bf16.h>

// ---------------------------------------------------------------------------
// CrossModalMultiHeadAttentionK for MI455X (gfx1250, wave32, WMMA bf16 path)
//
// Pipeline (all on `stream`):
//   1. prep_q : query + sine PE  -> bf16 rows [0..255] of fused [512 x 1600]
//   2. prep_k : padded key (+PE) -> bf16 [256 x 2116] (kpe) and raw (pkey)
//   3. gemm   : q  = (Wq @ qpe + bq) * 1/sqrt(32)          (f32 out)
//   4. gemm   : kk =  Wk @ kpe + bk                        (f32 out)
//   5. gemm   : v  =  Wv @ pkey + bv                       (f32 out)
//   6. attn   : per-channel 7x7 softmax-weighted sum -> bf16 ao
//   7. gemm   : vo =  Wo @ ao + bo -> f32 into d_out[vo] + bf16 rows[256..511]
//   8. gemm   : out = Wf @ concat(qpe, vo)  (K=512) -> f32 d_out[out]
//
// GEMM: 256 threads = 8 waves (4 along M x 2 along N), block tile 128x128,
// wave tile 32x64 (2x4 WMMA accumulators), BK=32, double-buffered LDS with a
// single workgroup barrier per K-step. Tile fills use a batched, branch-free
// global-load phase (B loads address-clamped: out-of-tile columns only feed
// output columns that are never stored) then a convert/LDS-store phase.
// ---------------------------------------------------------------------------

typedef float  v8f   __attribute__((ext_vector_type(8)));
typedef __bf16 v8bf  __attribute__((ext_vector_type(8)));
typedef __bf16 v16bf __attribute__((ext_vector_type(16)));

#define BM   128
#define BN   128
#define BK   32
#define LDA  40   // padded LDS row stride (bf16 elems) for A tile
#define LDB  40   // padded LDS row stride (bf16 elems) for B^T tile

// ---- sine positional encoding (analytic form of the reference) ------------
// dim_t[i] = 10000^(2*floor(i/2)/128); channel c<128 uses y, >=128 uses x.
__device__ __forceinline__ float pe_val(int c, float yn, float xn) {
  float base = (c < 128) ? yn : xn;
  int cc = c & 127;
  int j = cc >> 1;
  // 10000^{-j/64} = exp(-j * ln(10000)/64)
  float inv = __expf(-0.14391156831212787f * (float)j);
  float arg = base * inv;
  return (cc & 1) ? __cosf(arg) : __sinf(arg);
}

// 2*pi / (40 + 1e-6)  (cumsum of all-ones normalized by its last element)
#define PE_NORM (6.283185307179586f / 40.000001f)

// ---- 1: query + PE -> bf16 rows [0..255] of fused buffer -------------------
__global__ __launch_bounds__(256) void prep_q_kernel(
    const float* __restrict__ q, __bf16* __restrict__ fuse) {
  int idx = blockIdx.x * 256 + threadIdx.x;        // < 2*256*1600
  int hw = idx % 1600;
  int t  = idx / 1600;
  int c = t % 256, b = t / 256;
  int h = hw / 40, w = hw % 40;
  float yn = (float)(h + 1) * PE_NORM;
  float xn = (float)(w + 1) * PE_NORM;
  float v = q[idx] + pe_val(c, yn, xn);
  fuse[((long)b * 512 + c) * 1600 + hw] = (__bf16)v;
}

// ---- 2: padded key (46x46) -> pkey bf16 and key+PE bf16 --------------------
__global__ __launch_bounds__(256) void prep_k_kernel(
    const float* __restrict__ key, __bf16* __restrict__ kpe,
    __bf16* __restrict__ pkey) {
  int idx = blockIdx.x * 256 + threadIdx.x;        // < 2*256*2116
  int n = idx % 2116;
  int t = idx / 2116;
  int c = t % 256, b = t / 256;
  int ph = n / 46, pw = n % 46;
  bool inb = (ph >= 3) & (ph < 43) & (pw >= 3) & (pw < 43);
  float kv = inb ? key[((long)b * 256 + c) * 1600 + (ph - 3) * 40 + (pw - 3)]
                 : 0.f;
  // masked cumsum collapses to clamped counts; border rows/cols -> 0
  float yn = 0.f, xn = 0.f;
  if (pw >= 3 && pw < 43) {
    int cnt = (ph < 3) ? 0 : ((ph > 42 ? 42 : ph) - 2);
    yn = (float)cnt * PE_NORM;
  }
  if (ph >= 3 && ph < 43) {
    int cnt = (pw < 3) ? 0 : ((pw > 42 ? 42 : pw) - 2);
    xn = (float)cnt * PE_NORM;
  }
  pkey[idx] = (__bf16)kv;
  kpe[idx]  = (__bf16)(kv + pe_val(c, yn, xn));
}

// ---- 3-5,7,8: bf16 WMMA GEMM  C = (A*B + bias) * scale ---------------------
// A: f32 [256 x K] row-major (weights, converted to bf16 on LDS fill)
// B: bf16 [K x N] row-major per batch (stride bsB), N even; edge columns are
//    handled by address clamping (see below), stores are bounds-checked.
// Cf: optional f32 [256 x N] per batch; Cb: optional bf16 copy
__global__ __launch_bounds__(256) void gemm_bf16_kernel(
    const float* __restrict__ A, const __bf16* __restrict__ B,
    const float* __restrict__ bias, float* __restrict__ Cf,
    __bf16* __restrict__ Cb, int K, int N, long bsB, long bsC, long bsCb,
    float scale) {
  __shared__ __align__(16) __bf16 As[2][BM * LDA];
  __shared__ __align__(16) __bf16 Bs[2][BN * LDB];

  const int tid  = threadIdx.x;
  const int lane = tid & 31;
  const int wave = tid >> 5;
  const int wm = (wave >> 1) * 32;   // 4 waves along M : 128
  const int wn = (wave & 1) * 64;    // 2 waves along N : 128
  const int m0 = blockIdx.y * BM;
  const int n0 = blockIdx.x * BN;
  const long zB = (long)blockIdx.z * bsB;
  const int mlane = lane & 15;       // M (A) / N (B) index within fragment
  const int kh    = lane >> 4;       // K-half select

  v8f acc[2][4] = {};

  const int arow = tid >> 3;             // 0..31
  const int acg  = (tid & 7) * 4;        // 4-float column group
  const int bkr  = tid >> 6;             // 0..3
  const int bcol = (tid & 63) * 2;       // 0..126 (even)
  // Clamp B column so loads stay in-bounds; clamped (stale) values only feed
  // LDS columns >= N, whose outputs are never stored. N is even; bcol even,
  // so the u32 pair load at column nc is 4-byte aligned and fully in-bounds.
  const int nc = (n0 + bcol < N - 2) ? (n0 + bcol) : (N - 2);

  const int nk = K / BK;
  for (int kt = 0; kt < nk; ++kt) {
    const int k0  = kt * BK;
    const int buf = kt & 1;
    // prefetch next K tile (speculative, gfx1250 global_prefetch_b8)
    if (kt + 1 < nk) {
      __builtin_prefetch(A + (long)(m0 + (tid >> 1)) * K + k0 + BK, 0, 0);
      __builtin_prefetch(B + zB + (long)(k0 + BK + (tid >> 5)) * N + nc, 0, 0);
    }

    // ---- batched, branch-free global-load phase ----
    float4 af[4];
#pragma unroll
    for (int rr = 0; rr < 4; ++rr)
      af[rr] = *(const float4*)(A + (long)(m0 + arow + rr * 32) * K + k0 + acg);

    unsigned bb[8];                       // one aligned u32 = 2 bf16 per row
#pragma unroll
    for (int rr = 0; rr < 8; ++rr)
      bb[rr] = *(const unsigned*)(B + zB + (long)(k0 + bkr + rr * 4) * N + nc);

    // ---- convert / LDS-store phase ----
#pragma unroll
    for (int rr = 0; rr < 4; ++rr) {
      int r = arow + rr * 32;
      As[buf][r * LDA + acg + 0] = (__bf16)af[rr].x;
      As[buf][r * LDA + acg + 1] = (__bf16)af[rr].y;
      As[buf][r * LDA + acg + 2] = (__bf16)af[rr].z;
      As[buf][r * LDA + acg + 3] = (__bf16)af[rr].w;
    }
#pragma unroll
    for (int rr = 0; rr < 8; ++rr) {
      int k = bkr + rr * 4;
      __bf16 lo = __builtin_bit_cast(__bf16, (unsigned short)(bb[rr] & 0xffffu));
      __bf16 hi = __builtin_bit_cast(__bf16, (unsigned short)(bb[rr] >> 16));
      Bs[buf][(bcol    ) * LDB + k] = lo;
      Bs[buf][(bcol + 1) * LDB + k] = hi;
    }
    __syncthreads();   // single barrier per K-step (double-buffered LDS)

    // fragments per the CDNA5 16-bit A / B VGPR layouts (contiguous b128 LDS)
    v16bf a[2], b[4];
#pragma unroll
    for (int t = 0; t < 2; ++t) {
      const __bf16* ap = &As[buf][(wm + t * 16 + mlane) * LDA + kh * 8];
      union { v16bf v; v8bf h[2]; } ua;
      ua.h[0] = *(const v8bf*)(ap);        // K = kh*8 .. kh*8+7
      ua.h[1] = *(const v8bf*)(ap + 16);   // K = 16+kh*8 .. 16+kh*8+7
      a[t] = ua.v;
    }
#pragma unroll
    for (int t = 0; t < 4; ++t) {
      const __bf16* bp = &Bs[buf][(wn + t * 16 + mlane) * LDB + kh * 8];
      union { v16bf v; v8bf h[2]; } ub;
      ub.h[0] = *(const v8bf*)(bp);        // K = kh*8 .. kh*8+7
      ub.h[1] = *(const v8bf*)(bp + 16);   // K = 16+kh*8 .. 16+kh*8+7
      b[t] = ub.v;
    }
#pragma unroll
    for (int tm = 0; tm < 2; ++tm)
#pragma unroll
      for (int tn = 0; tn < 4; ++tn)
        acc[tm][tn] = __builtin_amdgcn_wmma_f32_16x16x32_bf16(
            false, a[tm], false, b[tn], (short)0, acc[tm][tn], false, false);
  }

  // epilogue: f32 C/D layout -> M = r + 8*(lane>>4), N = lane&15
  const long zC  = (long)blockIdx.z * bsC;
  const long zCb = (long)blockIdx.z * bsCb;
#pragma unroll
  for (int tm = 0; tm < 2; ++tm)
#pragma unroll
    for (int tn = 0; tn < 4; ++tn)
#pragma unroll
      for (int r = 0; r < 8; ++r) {
        int m = m0 + wm + tm * 16 + r + kh * 8;
        int n = n0 + wn + tn * 16 + mlane;
        if (n < N) {
          float v = acc[tm][tn][r];
          if (bias) v += bias[m];
          v *= scale;
          if (Cf) Cf[zC + (long)m * N + n] = v;
          if (Cb) Cb[zCb + (long)m * N + n] = (__bf16)v;
        }
      }
}

// ---- 6: per-channel 7x7 softmax attention ----------------------------------
__global__ __launch_bounds__(256) void attn_kernel(
    const float* __restrict__ qf, const float* __restrict__ kkf,
    const float* __restrict__ vf, __bf16* __restrict__ ao) {
  int idx = blockIdx.x * 256 + threadIdx.x;   // < 2*256*1600
  int hw = idx % 1600;
  int t  = idx / 1600;
  int c = t % 256, b = t / 256;
  int h = hw / 40, w = hw % 40;
  float q = qf[idx];
  long base = ((long)b * 256 + c) * 2116 + (long)h * 46 + w;
  float s[49];
  float mx = -3.4e38f;
#pragma unroll
  for (int dy = 0; dy < 7; ++dy)
#pragma unroll
    for (int dx = 0; dx < 7; ++dx) {
      float sv = q * kkf[base + dy * 46 + dx];
      s[dy * 7 + dx] = sv;
      mx = fmaxf(mx, sv);
    }
  float sum = 0.f, acc = 0.f;
#pragma unroll
  for (int i = 0; i < 49; ++i) {
    float e = __expf(s[i] - mx);
    sum += e;
    acc += e * vf[base + (i / 7) * 46 + (i % 7)];
  }
  ao[idx] = (__bf16)(acc / sum);
}

// ---------------------------------------------------------------------------
extern "C" void kernel_launch(void* const* d_in, const int* in_sizes, int n_in,
                              void* d_out, int out_size, void* d_ws,
                              size_t ws_size, hipStream_t stream) {
  const float* key   = (const float*)d_in[0];
  const float* query = (const float*)d_in[1];
  const float* Wq = (const float*)d_in[2];
  const float* bq = (const float*)d_in[3];
  const float* Wk = (const float*)d_in[4];
  const float* bk = (const float*)d_in[5];
  const float* Wv = (const float*)d_in[6];
  const float* bv = (const float*)d_in[7];
  const float* Wo = (const float*)d_in[8];
  const float* bo = (const float*)d_in[9];
  const float* Wf = (const float*)d_in[10];

  float* out = (float*)d_out;                 // (2,256,40,40)
  float* vo  = out + (size_t)2 * 256 * 1600;  // second output (2,256,40,40)

  char* ws = (char*)d_ws;
  __bf16* fuse = (__bf16*)ws; ws += (size_t)2 * 512 * 1600 * 2; // qpe | vo_bf16
  __bf16* kpe  = (__bf16*)ws; ws += (size_t)2 * 256 * 2116 * 2;
  __bf16* pkey = (__bf16*)ws; ws += (size_t)2 * 256 * 2116 * 2;
  float*  qf   = (float*) ws; ws += (size_t)2 * 256 * 1600 * 4;
  float*  kkf  = (float*) ws; ws += (size_t)2 * 256 * 2116 * 4;
  float*  vff  = (float*) ws; ws += (size_t)2 * 256 * 2116 * 4;
  __bf16* ao   = (__bf16*)ws;                                   // [2,256,1600]

  dim3 blk(256);

  prep_q_kernel<<<3200, blk, 0, stream>>>(query, fuse);
  prep_k_kernel<<<4232, blk, 0, stream>>>(key, kpe, pkey);

  // q = (Wq @ qpe + bq) / sqrt(32)
  gemm_bf16_kernel<<<dim3(13, 2, 2), blk, 0, stream>>>(
      Wq, fuse, bq, qf, nullptr, 256, 1600,
      (long)512 * 1600, (long)256 * 1600, 0, 0.17677669529663687f);
  // kk = Wk @ kpe + bk
  gemm_bf16_kernel<<<dim3(17, 2, 2), blk, 0, stream>>>(
      Wk, kpe, bk, kkf, nullptr, 256, 2116,
      (long)256 * 2116, (long)256 * 2116, 0, 1.f);
  // v = Wv @ pkey + bv
  gemm_bf16_kernel<<<dim3(17, 2, 2), blk, 0, stream>>>(
      Wv, pkey, bv, vff, nullptr, 256, 2116,
      (long)256 * 2116, (long)256 * 2116, 0, 1.f);

  attn_kernel<<<3200, blk, 0, stream>>>(qf, kkf, vff, ao);

  // vo = Wo @ ao + bo  -> f32 output + bf16 into fused rows [256..511]
  gemm_bf16_kernel<<<dim3(13, 2, 2), blk, 0, stream>>>(
      Wo, ao, bo, vo, fuse + (size_t)256 * 1600, 256, 1600,
      (long)256 * 1600, (long)256 * 1600, (long)512 * 1600, 1.f);
  // out = Wf @ concat(qpe, vo)   (K = 512, no bias)
  gemm_bf16_kernel<<<dim3(13, 2, 2), blk, 0, stream>>>(
      Wf, fuse, nullptr, out, nullptr, 512, 1600,
      (long)512 * 1600, (long)256 * 1600, 0, 1.f);
}
